// PerformerAttention_48859547959758
// MI455X (gfx1250) — compile-verified, matrix operand order checked by reading, not run on hardware
//
#include <hip/hip_runtime.h>

#define HD 64
#define RD 256
#define NN 4096
#define BHh 32
#define TNr 128
#define KEPS 1e-3f
#define QSCALE 0.35355339059327373f  // 64^(-1/4)

typedef _Float16 h4  __attribute__((ext_vector_type(4)));
typedef _Float16 h8  __attribute__((ext_vector_type(8)));
typedef _Float16 v16h __attribute__((ext_vector_type(16)));
typedef float    v4f __attribute__((ext_vector_type(4)));
typedef float    v8f __attribute__((ext_vector_type(8)));

// Load one 16x32 f16 WMMA A/B fragment from a row-major [row][K] f16 array.
// 16-bit A layout: lanes 0-15 hold K=[kb..kb+7] (VGPR0-3) and K=[kb+16..kb+23]
// (VGPR4-7); lanes 16-31 hold K+8 of each -> two 16B loads per lane.
__device__ __forceinline__ v16h load_frag(const _Float16* p, int row, int stride,
                                          int kbase, int lane) {
  const int ko = kbase + ((lane >> 4) << 3);
  const _Float16* rp = p + row * stride + ko;
  h8 lo = *(const h8*)(rp);
  h8 hi = *(const h8*)(rp + 16);
  return __builtin_shufflevector(lo, hi, 0,1,2,3,4,5,6,7,8,9,10,11,12,13,14,15);
}

__device__ __forceinline__ v8f wmma16(v16h a, v16h b, v8f c) {
  return __builtin_amdgcn_wmma_f32_16x16x32_f16(false, a, false, b, (short)0, c,
                                                false, false);
}

__global__ __launch_bounds__(256) void k_init(const float* __restrict__ proj,
                                              float* __restrict__ ctx,
                                              float* __restrict__ ksum,
                                              _Float16* __restrict__ projh) {
  int i = blockIdx.x * 256 + threadIdx.x;
  if (i < BHh * RD * HD) ctx[i] = 0.f;
  if (i < BHh * RD) ksum[i] = 0.f;
  if (i < RD * HD) projh[i] = (_Float16)proj[i];
}

// Pass 1: context[bh][r][d] += phi_k^T @ V ; ksum[bh][r] += sum_n phi_k
__global__ __launch_bounds__(256) void k_ctx(const float* __restrict__ K,
                                             const float* __restrict__ V,
                                             const float* __restrict__ mask,
                                             const _Float16* __restrict__ projh,
                                             float* __restrict__ ctx,
                                             float* __restrict__ ksum) {
  __shared__ _Float16 Kt[TNr * 72];        // [n][d], padded
  __shared__ _Float16 Vt[HD * 136];        // [d][n], padded (V transposed)
  __shared__ _Float16 phiT[RD * 136];      // [r][n], padded
  const int tid = threadIdx.x, lane = tid & 31, wv = tid >> 5;
  const int bh = blockIdx.y, b = bh >> 3, n0 = blockIdx.x * TNr;
  const float* Kb = K + ((size_t)bh * NN + n0) * HD;
  const float* Vb = V + ((size_t)bh * NN + n0) * HD;

  // stage tiles: K*scale*mask row-major f16; V*mask transposed f16
  for (int it = 0; it < 8; ++it) {
    int idx = tid + it * 256;              // 2048 float4 units
    int row = idx >> 4, c4 = (idx & 15) << 2;
    float m = mask[b * NN + n0 + row];
    v4f kq = *(const v4f*)(Kb + row * HD + c4);
    v4f vq = *(const v4f*)(Vb + row * HD + c4);
    h4 kh;
#pragma unroll
    for (int j = 0; j < 4; ++j) kh[j] = (_Float16)(kq[j] * (QSCALE * m));
    *(h4*)&Kt[row * 72 + c4] = kh;
#pragma unroll
    for (int j = 0; j < 4; ++j) Vt[(c4 + j) * 136 + row] = (_Float16)(vq[j] * m);
  }
  __syncthreads();

  // GEMM1: phi_k = relu(Kt @ proj^T)+eps. M=n (wave owns 16 rows), N=r, K=d.
  {
    v16h a0 = load_frag(Kt, wv * 16 + (lane & 15), 72, 0, lane);
    v16h a1 = load_frag(Kt, wv * 16 + (lane & 15), 72, 32, lane);
    for (int rt = 0; rt < 16; ++rt) {
      v16h b0 = load_frag(projh, rt * 16 + (lane & 15), HD, 0, lane);
      v16h b1 = load_frag(projh, rt * 16 + (lane & 15), HD, 32, lane);
      v8f c = {};
      c = wmma16(a0, b0, c);
      c = wmma16(a1, b1, c);
      h8 ph;
#pragma unroll
      for (int i = 0; i < 8; ++i) ph[i] = (_Float16)(fmaxf(c[i], 0.f) + KEPS);
      int r = rt * 16 + (lane & 15);                 // D-tile N index
      int nc = wv * 16 + ((lane >> 4) << 3);         // D-tile M base
      *(h8*)&phiT[r * 136 + nc] = ph;                // store transposed [r][n]
    }
  }
  __syncthreads();

  // ksum: one r-row per thread (256 threads == R), vectorized h8 reads
  {
    float s = 0.f;
    const _Float16* pr = &phiT[tid * 136];
#pragma unroll
    for (int n = 0; n < TNr; n += 8) {
      h8 ph = *(const h8*)(pr + n);
#pragma unroll
      for (int j = 0; j < 8; ++j) s += (float)ph[j];
    }
    atomicAdd(&ksum[bh * RD + tid], s);
  }

  // GEMM2: ctx += phiT(r,n) @ V(n,d). Wave owns r in [wv*32, wv*32+32), d full.
  {
    v8f acc[2][4];
#pragma unroll
    for (int i = 0; i < 2; ++i)
#pragma unroll
      for (int j = 0; j < 4; ++j) { v8f z = {}; acc[i][j] = z; }
    for (int kc = 0; kc < TNr; kc += 32) {
      v16h a0 = load_frag(phiT, wv * 32 + (lane & 15), 136, kc, lane);
      v16h a1 = load_frag(phiT, wv * 32 + 16 + (lane & 15), 136, kc, lane);
#pragma unroll
      for (int dt = 0; dt < 4; ++dt) {
        v16h bf = load_frag(Vt, dt * 16 + (lane & 15), 136, kc, lane);
        acc[0][dt] = wmma16(a0, bf, acc[0][dt]);
        acc[1][dt] = wmma16(a1, bf, acc[1][dt]);
      }
    }
    float* cb = ctx + (size_t)bh * RD * HD;
#pragma unroll
    for (int i = 0; i < 2; ++i)
#pragma unroll
      for (int dt = 0; dt < 4; ++dt)
#pragma unroll
        for (int e = 0; e < 8; ++e) {
          int r = wv * 32 + i * 16 + ((lane >> 4) << 3) + e;
          int d = dt * 16 + (lane & 15);
          atomicAdd(&cb[r * HD + d], acc[i][dt][e]);
        }
  }
}

// Pass 2: out = (phi_q @ ctx) * (1 / (phi_q . ksum))
__global__ __launch_bounds__(256) void k_out(const float* __restrict__ Q,
                                             const _Float16* __restrict__ projh,
                                             const float* __restrict__ ctx,
                                             const float* __restrict__ ksum,
                                             float* __restrict__ out) {
  __shared__ _Float16 Qt[TNr * 72];        // [n][d]
  __shared__ _Float16 ctxT[HD * 264];      // [d][r]
  __shared__ _Float16 phiQ[TNr * 264];     // [n][r] row-major
  __shared__ float ksumL[RD];
  __shared__ float dinv[TNr];
  const int tid = threadIdx.x, lane = tid & 31, wv = tid >> 5;
  const int bh = blockIdx.y, n0 = blockIdx.x * TNr;
  const float* Qb = Q + ((size_t)bh * NN + n0) * HD;
  const float* cb = ctx + (size_t)bh * RD * HD;

  ksumL[tid] = ksum[bh * RD + tid];        // 256 threads == R
  for (int it = 0; it < 8; ++it) {
    int idx = tid + it * 256;
    int row = idx >> 4, c4 = (idx & 15) << 2;
    v4f qv = *(const v4f*)(Qb + row * HD + c4);
    h4 qh;
#pragma unroll
    for (int j = 0; j < 4; ++j) qh[j] = (_Float16)(qv[j] * QSCALE);
    *(h4*)&Qt[row * 72 + c4] = qh;
  }
  for (int it = 0; it < 16; ++it) {
    int idx = tid + it * 256;              // 4096 float4 of ctx
    int r = idx >> 4, c4 = (idx & 15) << 2;
    v4f cv = *(const v4f*)(cb + r * HD + c4);
#pragma unroll
    for (int j = 0; j < 4; ++j) ctxT[(c4 + j) * 264 + r] = (_Float16)cv[j];
  }
  __syncthreads();

  // GEMM1 as proj @ Qt^T: M=r, N=n (wave owns 16 n-cols), K=d.
  // D-tile then stores ROW-major into phiQ[n][r] with one b128 per lane.
  {
    v16h b0 = load_frag(Qt, wv * 16 + (lane & 15), 72, 0, lane);
    v16h b1 = load_frag(Qt, wv * 16 + (lane & 15), 72, 32, lane);
    for (int rt = 0; rt < 16; ++rt) {
      v16h a0 = load_frag(projh, rt * 16 + (lane & 15), HD, 0, lane);
      v16h a1 = load_frag(projh, rt * 16 + (lane & 15), HD, 32, lane);
      v8f c = {};
      c = wmma16(a0, b0, c);
      c = wmma16(a1, b1, c);
      h8 ph;
#pragma unroll
      for (int i = 0; i < 8; ++i) ph[i] = (_Float16)(fmaxf(c[i], 0.f) + KEPS);
      int n = wv * 16 + (lane & 15);
      int rc = rt * 16 + ((lane >> 4) << 3);
      *(h8*)&phiQ[n * 264 + rc] = ph;
    }
  }
  __syncthreads();

  if (tid < TNr) {
    const _Float16* pr = &phiQ[tid * 264];
    float s = 0.f;
#pragma unroll
    for (int r = 0; r < RD; r += 8) {
      h8 ph = *(const h8*)(pr + r);
#pragma unroll
      for (int j = 0; j < 8; ++j) s += (float)ph[j] * ksumL[r + j];
    }
    dinv[tid] = 1.f / s;
  }
  __syncthreads();

  // GEMM2: out[n][d]. Wave owns n-tile wv; K=r over 256.
  {
    v8f acc[4];
#pragma unroll
    for (int j = 0; j < 4; ++j) { v8f z = {}; acc[j] = z; }
    for (int kc = 0; kc < RD; kc += 32) {
      v16h a = load_frag(phiQ, wv * 16 + (lane & 15), 264, kc, lane);
#pragma unroll
      for (int dt = 0; dt < 4; ++dt) {
        v16h bf = load_frag(ctxT, dt * 16 + (lane & 15), 264, kc, lane);
        acc[dt] = wmma16(a, bf, acc[dt]);
      }
    }
    float* ob = out + ((size_t)bh * NN + n0) * HD;
#pragma unroll
    for (int dt = 0; dt < 4; ++dt)
#pragma unroll
      for (int e = 0; e < 8; ++e) {
        int n = wv * 16 + ((lane >> 4) << 3) + e;
        int d = dt * 16 + (lane & 15);
        ob[n * HD + d] = acc[dt][e] * dinv[n];
      }
  }
}

extern "C" void kernel_launch(void* const* d_in, const int* in_sizes, int n_in,
                              void* d_out, int out_size, void* d_ws, size_t ws_size,
                              hipStream_t stream) {
  const float* Q    = (const float*)d_in[0];
  const float* K    = (const float*)d_in[1];
  const float* V    = (const float*)d_in[2];
  const float* mask = (const float*)d_in[3];
  const float* proj = (const float*)d_in[4];
  float* out = (float*)d_out;

  float* ctx  = (float*)d_ws;                       // 32*256*64 f32 = 2 MB
  float* ksum = ctx + BHh * RD * HD;                // 32*256 f32
  _Float16* projh = (_Float16*)(ksum + BHh * RD);   // 256*64 f16

  k_init<<<dim3((BHh * RD * HD + 255) / 256), 256, 0, stream>>>(proj, ctx, ksum, projh);
  dim3 grid(NN / TNr, BHh);
  k_ctx<<<grid, 256, 0, stream>>>(K, V, mask, projh, ctx, ksum);
  k_out<<<grid, 256, 0, stream>>>(Q, projh, ctx, ksum, out);
}
